// EfficientAttention_91182155694184
// MI455X (gfx1250) — matrix operand order repaired and checked by reading, not compiled, not run on hardware
//
#include <hip/hip_runtime.h>

typedef __attribute__((ext_vector_type(16))) _Float16 v16h;
typedef __attribute__((ext_vector_type(8)))  float    v8f;

#define ATT_N 4096
#define ATT_D 64
#define HEADS 4
#define BATCH 2
#define L2E 1.44269504088896340736f

union V16U { v16h v; uint4 q[2]; };

__device__ __forceinline__ v16h ld2x8(const _Float16* p0, const _Float16* p1) {
  V16U u;
  u.q[0] = *(const uint4*)p0;
  u.q[1] = *(const uint4*)p1;
  return u.v;
}

__device__ __forceinline__ v8f wmma_f16(v16h a, v16h b, v8f c) {
  // D = A(16x32 f16) x B(32x16 f16) + C(16x16 f32)
  return __builtin_amdgcn_wmma_f32_16x16x32_f16(false, a, false, b, (short)0, c,
                                                false, false);
}

__device__ __forceinline__ float rmax16(float t) {
  t = fmaxf(t, __shfl_xor(t, 1, 32));
  t = fmaxf(t, __shfl_xor(t, 2, 32));
  t = fmaxf(t, __shfl_xor(t, 4, 32));
  t = fmaxf(t, __shfl_xor(t, 8, 32));
  return t;
}
__device__ __forceinline__ float rsum16(float t) {
  t += __shfl_xor(t, 1, 32);
  t += __shfl_xor(t, 2, 32);
  t += __shfl_xor(t, 4, 32);
  t += __shfl_xor(t, 8, 32);
  return t;
}

// Async global->LDS copy of one 16B chunk per lane (CDNA5 async path, ASYNCcnt)
__device__ __forceinline__ void async_cp16(const _Float16* g, _Float16* l) {
  const unsigned lds = (unsigned)(uintptr_t)l;
  const unsigned long long ga = (unsigned long long)(uintptr_t)g;
  asm volatile("global_load_async_to_lds_b128 %0, %1, off"
               :: "v"(lds), "v"(ga) : "memory");
}

// ---------------- Kernel 1: weight conversion (fold softmax scale into Q rows)
__global__ __launch_bounds__(256) void prep_weights(const float* __restrict__ w_qkv,
                                                    const float* __restrict__ w_out,
                                                    _Float16* __restrict__ wqh,
                                                    _Float16* __restrict__ woh) {
  const int i = blockIdx.x * 256 + threadIdx.x;
  if (i < 768 * 64) {
    const int o = i >> 6;
    const float s = (o < 256) ? 0.125f : 1.0f;  // dim_head^-0.5 = 1/8 on Q
    wqh[i] = (_Float16)(w_qkv[i] * s);
  }
  if (i < 64 * 256) woh[i] = (_Float16)w_out[i];
}

// ---------------- Kernel 2: x[b,c,n] (f32) -> xt[b,n,c] (f16), LDS tile transpose
__global__ __launch_bounds__(256) void transpose_x(const float* __restrict__ x,
                                                   _Float16* __restrict__ xt) {
  __shared__ float tile[64][65];
  const int b = blockIdx.y;
  const int nb = blockIdx.x * 64;
  const int tlo = threadIdx.x & 63;
  const int thi = threadIdx.x >> 6;
#pragma unroll
  for (int i = 0; i < 16; ++i) {
    const int c = thi + i * 4;
    tile[c][tlo] = x[((size_t)(b * 64 + c)) * ATT_N + nb + tlo];
  }
  __syncthreads();
#pragma unroll
  for (int i = 0; i < 16; ++i) {
    const int n = thi + i * 4;
    xt[((size_t)b * ATT_N + nb + n) * 64 + tlo] = (_Float16)tile[tlo][n];
  }
}

// ---------------- Kernel 3: QKV projection via WMMA
// qkv[o,n] = sum_c wqh[o,c] * xt[n,c]; scatter to Qt[bh][n][d], Kt[bh][m][d], V[bh][d][m]
__global__ __launch_bounds__(128) void qkv_gemm(const _Float16* __restrict__ wqh,
                                                const _Float16* __restrict__ xt,
                                                _Float16* __restrict__ Qt,
                                                _Float16* __restrict__ Kt,
                                                _Float16* __restrict__ Vv) {
  const int b = blockIdx.z;
  const int wave = threadIdx.x >> 5;
  const int lane = threadIdx.x & 31;
  const int l16 = lane & 15;
  const int lhalf = lane >> 4;
  const int klo = lhalf * 16;
  const int otile = blockIdx.y * 64 + wave * 16;
  const int nb0 = blockIdx.x * 64;

  // A fragments: rows o = otile..otile+15, K = c (two 32-chunks)
  const _Float16* wrow = wqh + (size_t)(otile + l16) * 64;
  const v16h a0 = ld2x8(wrow + klo,      wrow + klo + 8);
  const v16h a1 = ld2x8(wrow + 32 + klo, wrow + 32 + klo + 8);

  // Wave-uniform destination selection (scalarized so stores stay branch-free)
  const int seg   = __builtin_amdgcn_readfirstlane(otile >> 8);  // 0=q,1=k,2=v
  const int head  = __builtin_amdgcn_readfirstlane((otile >> 6) & 3);
  const int dbase = otile & 63;
  const int bh = b * HEADS + head;

  _Float16* dbaseptr;
  size_t sn, sd;  // dst = dbaseptr + n*sn + d*sd
  if (seg == 0)      { dbaseptr = Qt + (size_t)bh * ATT_N * ATT_D; sn = ATT_D; sd = 1; }
  else if (seg == 1) { dbaseptr = Kt + (size_t)bh * ATT_N * ATT_D; sn = ATT_D; sd = 1; }
  else               { dbaseptr = Vv + (size_t)bh * ATT_D * ATT_N; sn = 1; sd = ATT_N; }

#pragma unroll
  for (int s = 0; s < 4; ++s) {
    const int nb = nb0 + 16 * s;
    // B fragment: col n = nb+l16, K rows c contiguous in xt[b][n][c]
    const _Float16* xc = xt + ((size_t)b * ATT_N + nb + l16) * 64 + lhalf * 16;
    v16h b0 = ld2x8(xc,      xc + 8);
    v16h b1 = ld2x8(xc + 32, xc + 40);
    v8f c = {0, 0, 0, 0, 0, 0, 0, 0};
    c = wmma_f16(a0, b0, c);
    c = wmma_f16(a1, b1, c);
#pragma unroll
    for (int j = 0; j < 8; ++j) {
      const int d = dbase + j + lhalf * 8;   // C layout: row = j (+8 hi half)
      const int n = nb + l16;                //           col = lane % 16
      dbaseptr[(size_t)n * sn + (size_t)d * sd] = (_Float16)c[j];
    }
  }
}

// ---------------- Kernel 4: flash attention
// Block = 4 waves, each wave owns 16 query rows. Per iteration: 64 keys.
// K/V tiles staged cooperatively via global_load_async_to_lds_b128 (double-buffered).
#define KPITCH 72   // padded LDS pitch in halves (144B = 36 dwords, spreads banks)
__global__ __launch_bounds__(128) void flash_attn(const _Float16* __restrict__ Qt,
                                                  const _Float16* __restrict__ Kt,
                                                  const _Float16* __restrict__ Vv,
                                                  _Float16* __restrict__ At) {
  const int bh = blockIdx.y;
  const int b = bh >> 2;
  const int head = bh & 3;
  const int tid = threadIdx.x;
  const int wave = tid >> 5;
  const int lane = tid & 31;
  const int l16 = lane & 15;
  const int lhalf = lane >> 4;
  const int klo = lhalf * 16;
  const int qbase = blockIdx.x * 64 + wave * 16;

  const _Float16* Qbh = Qt + (size_t)bh * ATT_N * ATT_D;
  const _Float16* Kbh = Kt + (size_t)bh * ATT_N * ATT_D;
  const _Float16* Vbh = Vv + (size_t)bh * ATT_D * ATT_N;

  __shared__ alignas(16) _Float16 Ktile[2][64][KPITCH];
  __shared__ alignas(16) _Float16 Vtile[2][64][KPITCH];
  __shared__ alignas(16) _Float16 pstage[4][16][KPITCH];

  // Q A-fragments, resident for the whole key sweep
  const _Float16* qrow = Qbh + (size_t)(qbase + l16) * ATT_D;
  const v16h qa0 = ld2x8(qrow + klo,      qrow + klo + 8);
  const v16h qa1 = ld2x8(qrow + 32 + klo, qrow + 32 + klo + 8);

  v8f O0 = {0, 0, 0, 0, 0, 0, 0, 0};
  v8f O1 = {0, 0, 0, 0, 0, 0, 0, 0};
  v8f O2 = {0, 0, 0, 0, 0, 0, 0, 0};
  v8f O3 = {0, 0, 0, 0, 0, 0, 0, 0};
  float mrow[8], lrow[8];
#pragma unroll
  for (int j = 0; j < 8; ++j) { mrow[j] = -3.0e30f; lrow[j] = 0.0f; }

  // Cooperative async stage of a 64x64 K tile and 64x64 V tile (8KB each):
  // 512 16B-chunks per tile, 128 threads x 4 chunks, 8 async ops per wave.
  auto stage = [&](int buf, int mb) {
#pragma unroll
    for (int i = 0; i < 4; ++i) {
      const int id = tid + 128 * i;
      const int r = id >> 3;
      const int c = (id & 7) * 8;
      async_cp16(Kbh + (size_t)(mb + r) * ATT_D + c, &Ktile[buf][r][c]);
      async_cp16(Vbh + (size_t)r * ATT_N + mb + c, &Vtile[buf][r][c]);
    }
  };

  stage(0, 0);

  for (int it = 0; it < ATT_N / 64; ++it) {
    const int buf = it & 1;
    if (it + 1 < ATT_N / 64) {
      stage(buf ^ 1, (it + 1) * 64);
      asm volatile("s_wait_asynccnt 0x8" ::: "memory");  // current tile done
    } else {
      asm volatile("s_wait_asynccnt 0x0" ::: "memory");
    }
    __syncthreads();  // publish staged tile across waves

    // S = Q * K^T for four 16-col key tiles (scale pre-folded into Q)
    v8f S0 = {0, 0, 0, 0, 0, 0, 0, 0};
    v8f S1 = {0, 0, 0, 0, 0, 0, 0, 0};
    v8f S2 = {0, 0, 0, 0, 0, 0, 0, 0};
    v8f S3 = {0, 0, 0, 0, 0, 0, 0, 0};
#pragma unroll
    for (int mt = 0; mt < 4; ++mt) {
      const _Float16* kr = &Ktile[buf][16 * mt + l16][lhalf * 16];
      v16h bk0 = ld2x8(kr,      kr + 8);
      v16h bk1 = ld2x8(kr + 32, kr + 40);
      if (mt == 0)      { S0 = wmma_f16(qa0, bk0, S0); S0 = wmma_f16(qa1, bk1, S0); }
      else if (mt == 1) { S1 = wmma_f16(qa0, bk0, S1); S1 = wmma_f16(qa1, bk1, S1); }
      else if (mt == 2) { S2 = wmma_f16(qa0, bk0, S2); S2 = wmma_f16(qa1, bk1, S2); }
      else              { S3 = wmma_f16(qa0, bk0, S3); S3 = wmma_f16(qa1, bk1, S3); }
    }

    // Online softmax over these 64 keys; a row lives across a 16-lane half
#pragma unroll
    for (int j = 0; j < 8; ++j) {
      float t = fmaxf(fmaxf(S0[j], S1[j]), fmaxf(S2[j], S3[j]));
      t = rmax16(t);
      const float mn = fmaxf(mrow[j], t);
      const float corr = __builtin_exp2f((mrow[j] - mn) * L2E);
      const float p0 = __builtin_exp2f((S0[j] - mn) * L2E);
      const float p1 = __builtin_exp2f((S1[j] - mn) * L2E);
      const float p2 = __builtin_exp2f((S2[j] - mn) * L2E);
      const float p3 = __builtin_exp2f((S3[j] - mn) * L2E);
      lrow[j] = lrow[j] * corr + rsum16(((p0 + p1) + (p2 + p3)));
      mrow[j] = mn;
      O0[j] *= corr; O1[j] *= corr; O2[j] *= corr; O3[j] *= corr;
      const int r = j + lhalf * 8;
      pstage[wave][r][l16]      = (_Float16)p0;
      pstage[wave][r][16 + l16] = (_Float16)p1;
      pstage[wave][r][32 + l16] = (_Float16)p2;
      pstage[wave][r][48 + l16] = (_Float16)p3;
    }
    // intra-wave LDS store->load ordering is guaranteed (LDS in-order per wave)

    // P as two 16x32 A-fragments
    const _Float16* pr = &pstage[wave][l16][0];
    const v16h ap0 = ld2x8(pr + klo,      pr + klo + 8);
    const v16h ap1 = ld2x8(pr + 32 + klo, pr + 32 + klo + 8);

    // O += P * V^T
#pragma unroll
    for (int t = 0; t < 4; ++t) {
      const _Float16* vr = &Vtile[buf][16 * t + l16][lhalf * 16];
      v16h bv0 = ld2x8(vr,      vr + 8);
      v16h bv1 = ld2x8(vr + 32, vr + 40);
      if (t == 0)      { O0 = wmma_f16(ap0, bv0, O0); O0 = wmma_f16(ap1, bv1, O0); }
      else if (t == 1) { O1 = wmma_f16(ap0, bv0, O1); O1 = wmma_f16(ap1, bv1, O1); }
      else if (t == 2) { O2 = wmma_f16(ap0, bv0, O2); O2 = wmma_f16(ap1, bv1, O2); }
      else             { O3 = wmma_f16(ap0, bv0, O3); O3 = wmma_f16(ap1, bv1, O3); }
    }

    __syncthreads();  // all waves done reading before next tile overwrites buffer
  }

  // Normalize and stage transposed attention output At[b][n][head*64+d] (f16)
  _Float16* abase = At + ((size_t)b * ATT_N + qbase) * 256 + head * 64;
#pragma unroll
  for (int j = 0; j < 8; ++j) {
    const float inv = 1.0f / lrow[j];
    const int r = j + lhalf * 8;
    _Float16* dst = abase + (size_t)r * 256 + l16;
    dst[0]  = (_Float16)(O0[j] * inv);
    dst[16] = (_Float16)(O1[j] * inv);
    dst[32] = (_Float16)(O2[j] * inv);
    dst[48] = (_Float16)(O3[j] * inv);
  }
}

// ---------------- Kernel 5: output projection via WMMA (+bias), f32 out
__global__ __launch_bounds__(128) void proj_out(const _Float16* __restrict__ woh,
                                                const _Float16* __restrict__ At,
                                                const float* __restrict__ b_out,
                                                float* __restrict__ out) {
  const int b = blockIdx.y;
  const int nb = blockIdx.x * 16;
  const int wave = threadIdx.x >> 5;
  const int lane = threadIdx.x & 31;
  const int l16 = lane & 15;
  const int lhalf = lane >> 4;
  const int klo = lhalf * 16;
  const int ob = wave * 16;

  const _Float16* wrow = woh + (size_t)(ob + l16) * 256;
  const _Float16* arow = At + ((size_t)b * ATT_N + nb + l16) * 256 + lhalf * 16;

  v8f acc = {0, 0, 0, 0, 0, 0, 0, 0};
#pragma unroll
  for (int k = 0; k < 8; ++k) {
    v16h a  = ld2x8(wrow + 32 * k + klo, wrow + 32 * k + klo + 8);
    v16h bb = ld2x8(arow + 32 * k,       arow + 32 * k + 8);
    acc = wmma_f16(a, bb, acc);
  }
#pragma unroll
  for (int j = 0; j < 8; ++j) {
    const int o = ob + j + lhalf * 8;
    out[((size_t)(b * 64 + o)) * ATT_N + nb + l16] = acc[j] + b_out[o];
  }
}

extern "C" void kernel_launch(void* const* d_in, const int* in_sizes, int n_in,
                              void* d_out, int out_size, void* d_ws, size_t ws_size,
                              hipStream_t stream) {
  (void)in_sizes; (void)n_in; (void)out_size; (void)ws_size;
  const float* x     = (const float*)d_in[0];
  const float* w_qkv = (const float*)d_in[1];
  const float* w_out = (const float*)d_in[2];
  const float* b_out = (const float*)d_in[3];
  float* out = (float*)d_out;

  char* ws = (char*)d_ws;
  size_t off = 0;
  auto carve = [&](size_t bytes) -> char* {
    char* p = ws + off;
    off += (bytes + 255) & ~(size_t)255;
    return p;
  };
  _Float16* wqh = (_Float16*)carve((size_t)768 * 64 * 2);
  _Float16* woh = (_Float16*)carve((size_t)64 * 256 * 2);
  _Float16* xt  = (_Float16*)carve((size_t)BATCH * ATT_N * ATT_D * 2);
  _Float16* Qt  = (_Float16*)carve((size_t)8 * ATT_N * ATT_D * 2);
  _Float16* Kt  = (_Float16*)carve((size_t)8 * ATT_N * ATT_D * 2);
  _Float16* Vv  = (_Float16*)carve((size_t)8 * ATT_N * ATT_D * 2);
  _Float16* At  = (_Float16*)carve((size_t)BATCH * ATT_N * 256 * 2);

  prep_weights<<<192, 256, 0, stream>>>(w_qkv, w_out, wqh, woh);
  transpose_x<<<dim3(64, BATCH), 256, 0, stream>>>(x, xt);
  qkv_gemm<<<dim3(64, 12, BATCH), 128, 0, stream>>>(wqh, xt, Qt, Kt, Vv);
  flash_attn<<<dim3(64, 8), 128, 0, stream>>>(Qt, Kt, Vv, At);
  proj_out<<<dim3(256, BATCH), 128, 0, stream>>>(woh, At, b_out, out);
}
